// _MHA_9809705304213
// MI455X (gfx1250) — matrix operand order repaired and checked by reading, not compiled, not run on hardware
//
#include <hip/hip_runtime.h>
#include <stdint.h>

typedef __attribute__((ext_vector_type(16))) _Float16 v16h;
typedef __attribute__((ext_vector_type(8)))  float    v8f;

union HFrag {
  v16h v;
  uint32_t u[8];
  _Float16 h[16];
};

#define WMMA_F16(A, B, C) \
  __builtin_amdgcn_wmma_f32_16x16x32_f16(false, (A), false, (B), (short)0, (C), false, false)

// A-matrix (16x32 f16) half-index -> K index (contiguous, matches B fills):
// lanes 0-15: V0..7 -> K 0..15 ; lanes 16-31: V0..7 -> K 16..31
static __device__ __forceinline__ int a_kd(int v, int hi) {
  return hi * 16 + 2 * v;
}

// ---------------------------------------------------------------------------
// Kernel 1: QKV GEMM.  x[4096,768] (f32) @ qkv_w[768,2304] (f32) + qkv_b.
// Writes f16: q (scaled by 1/8) [B,H,N,64], k [B,H,N,64], v transposed [B,H,64,N].
// ---------------------------------------------------------------------------
__global__ __launch_bounds__(256) void qkv_kernel(
    const float* __restrict__ x, const float* __restrict__ w,
    const float* __restrict__ bias,
    _Float16* __restrict__ qb, _Float16* __restrict__ kbuf,
    _Float16* __restrict__ vtb) {
  __shared__ _Float16 As[128 * 36];   // [row][k], stride 36 keeps dword alignment
  __shared__ _Float16 Bs[128 * 36];   // [n][k]   (B tile stored k-contiguous)
  const int tid  = threadIdx.x;
  const int lane = tid & 31, wave = tid >> 5;
  const int wm = wave >> 2, wn = wave & 3;      // 2x4 wave grid -> 64x32 per wave
  const int l16 = lane & 15, hi = lane >> 4;
  const int m0 = blockIdx.y * 128, n0 = blockIdx.x * 128;

  v8f acc[4][2];
#pragma unroll
  for (int mi = 0; mi < 4; ++mi)
#pragma unroll
    for (int ni = 0; ni < 2; ++ni)
#pragma unroll
      for (int r = 0; r < 8; ++r) acc[mi][ni][r] = 0.0f;

  for (int k0 = 0; k0 < 768; k0 += 32) {
    __syncthreads();
    // A tile: 128x32 f32 -> f16
#pragma unroll
    for (int it = 0; it < 4; ++it) {
      const int idx = tid + it * 256;          // 0..1023 float4 chunks
      const int row = idx >> 3, kc = (idx & 7) << 2;
      const float4 f = *(const float4*)(x + (size_t)(m0 + row) * 768 + k0 + kc);
      _Float16* dst = As + row * 36 + kc;
      dst[0] = (_Float16)f.x; dst[1] = (_Float16)f.y;
      dst[2] = (_Float16)f.z; dst[3] = (_Float16)f.w;
    }
    // B tile: 32x128 f32 -> f16, stored transposed [n][k]
#pragma unroll
    for (int it = 0; it < 4; ++it) {
      const int idx = tid + it * 256;
      const int kk = idx >> 5, nc = (idx & 31) << 2;
      const float4 f = *(const float4*)(w + (size_t)(k0 + kk) * 2304 + n0 + nc);
      Bs[(nc + 0) * 36 + kk] = (_Float16)f.x;
      Bs[(nc + 1) * 36 + kk] = (_Float16)f.y;
      Bs[(nc + 2) * 36 + kk] = (_Float16)f.z;
      Bs[(nc + 3) * 36 + kk] = (_Float16)f.w;
    }
    __syncthreads();

    HFrag a[4], b[2];
#pragma unroll
    for (int mi = 0; mi < 4; ++mi) {
      const uint32_t* p = (const uint32_t*)(As + (wm * 64 + mi * 16 + l16) * 36);
#pragma unroll
      for (int v = 0; v < 8; ++v) a[mi].u[v] = p[a_kd(v, hi) >> 1];
    }
#pragma unroll
    for (int ni = 0; ni < 2; ++ni) {
      const uint32_t* p = (const uint32_t*)(Bs + (wn * 32 + ni * 16 + l16) * 36);
#pragma unroll
      for (int v = 0; v < 8; ++v) b[ni].u[v] = p[v + hi * 8];  // K = 2v+16*hi
    }
#pragma unroll
    for (int mi = 0; mi < 4; ++mi)
#pragma unroll
      for (int ni = 0; ni < 2; ++ni)
        acc[mi][ni] = WMMA_F16(a[mi].v, b[ni].v, acc[mi][ni]);
  }

  // Epilogue: bias, split q/k/v, scatter to attention layouts (f16).
#pragma unroll
  for (int mi = 0; mi < 4; ++mi) {
#pragma unroll
    for (int ni = 0; ni < 2; ++ni) {
      const int col = n0 + wn * 32 + ni * 16 + l16;  // 0..2303
      const int sel = col / 768;                      // 0=q 1=k 2=v (16 | 768)
      const int c = col - sel * 768;
      const int head = c >> 6, d = c & 63;
      const float bv = bias[col];
#pragma unroll
      for (int r = 0; r < 8; ++r) {
        const int row = m0 + wm * 64 + mi * 16 + r + hi * 8;  // 0..4095
        const int bidx = row >> 10, nn = row & 1023;
        const float val = acc[mi][ni][r] + bv;
        const size_t nh = (size_t)(bidx * 12 + head);
        if (sel == 0)      qb  [(nh * 1024 + nn) * 64 + d] = (_Float16)(val * 0.125f);
        else if (sel == 1) kbuf[(nh * 1024 + nn) * 64 + d] = (_Float16)val;
        else               vtb [(nh * 64 + d) * 1024 + nn] = (_Float16)val;
      }
    }
  }
}

// ---------------------------------------------------------------------------
// Kernel 2: flash attention, one wave per 16 query rows, 32 keys per step.
// ---------------------------------------------------------------------------
__global__ __launch_bounds__(128) void attn_kernel(
    const _Float16* __restrict__ qb, const _Float16* __restrict__ kb,
    const _Float16* __restrict__ vtb, const float* __restrict__ abias,
    _Float16* __restrict__ aout) {
  __shared__ _Float16 plds[4][16 * 36];   // per-wave P transpose staging
  const int tid = threadIdx.x;
  const int lane = tid & 31, wave = tid >> 5;
  const int l16 = lane & 15, hi = lane >> 4;
  const int bh = blockIdx.y;              // 0..47 = b*12+h
  const int head = bh % 12, batch = bh / 12;
  const int qbase = blockIdx.x * 64 + wave * 16;

  const _Float16* qp = qb + (size_t)bh * (1024 * 64);
  const _Float16* kp = kb + (size_t)bh * (1024 * 64);
  const _Float16* vp = vtb + (size_t)bh * (1024 * 64);
  const float*    bp = abias + (size_t)head * 1024 * 1024;

  // Q 16x64 -> two 16x32 A-fragments (scale already folded in).
  HFrag qf[2];
  {
    const int row = qbase + l16;
#pragma unroll
    for (int c = 0; c < 2; ++c) {
      const uint32_t* p = (const uint32_t*)(qp + (size_t)row * 64 + 32 * c);
#pragma unroll
      for (int v = 0; v < 8; ++v) qf[c].u[v] = p[a_kd(v, hi) >> 1];
    }
  }

  v8f acc[4];
  float m[8], l[8];
#pragma unroll
  for (int t = 0; t < 4; ++t)
#pragma unroll
    for (int r = 0; r < 8; ++r) acc[t][r] = 0.0f;
#pragma unroll
  for (int r = 0; r < 8; ++r) { m[r] = -1e30f; l[r] = 0.0f; }

  _Float16* pl = plds[wave];

  for (int j = 0; j < 1024; j += 32) {
    // Prefetch next key block's K rows and V^T rows (global_prefetch_b8).
    if (j + 32 < 1024) {
      __builtin_prefetch((const void*)(kp + (size_t)(j + 32 + lane) * 64), 0, 1);
      __builtin_prefetch((const void*)(vp + (size_t)lane * 1024 + j + 32), 0, 1);
      __builtin_prefetch((const void*)(vp + (size_t)(32 + lane) * 1024 + j + 32), 0, 1);
    }

    // S = Q K^T (+bias): two 16-key subtiles, 2 WMMAs each (HD=64 = 2*32).
    v8f s[2];
#pragma unroll
    for (int t = 0; t < 2; ++t) {
      HFrag kf0, kf1;
      const int key = j + t * 16 + l16;
      const uint32_t* p0 = (const uint32_t*)(kp + (size_t)key * 64);
      const uint32_t* p1 = p0 + 16;
#pragma unroll
      for (int v = 0; v < 8; ++v) { kf0.u[v] = p0[v + hi * 8]; kf1.u[v] = p1[v + hi * 8]; }
      v8f z;
#pragma unroll
      for (int r = 0; r < 8; ++r) z[r] = 0.0f;
      z = WMMA_F16(qf[0].v, kf0.v, z);
      z = WMMA_F16(qf[1].v, kf1.v, z);
      const int col = j + t * 16 + l16;
#pragma unroll
      for (int r = 0; r < 8; ++r)
        z[r] += bp[(size_t)(qbase + r + hi * 8) * 1024 + col];
      s[t] = z;
    }

    // Online softmax: reductions across the 16-lane group that shares a row.
    float bm[8];
#pragma unroll
    for (int r = 0; r < 8; ++r) bm[r] = fmaxf(s[0][r], s[1][r]);
#pragma unroll
    for (int mask = 1; mask <= 8; mask <<= 1)
#pragma unroll
      for (int r = 0; r < 8; ++r) bm[r] = fmaxf(bm[r], __shfl_xor(bm[r], mask, 32));

    float corr[8], rs[8];
#pragma unroll
    for (int r = 0; r < 8; ++r) {
      const float mn = fmaxf(m[r], bm[r]);
      corr[r] = __expf(m[r] - mn);
      s[0][r] = __expf(s[0][r] - mn);
      s[1][r] = __expf(s[1][r] - mn);
      rs[r] = s[0][r] + s[1][r];
      m[r] = mn;
    }
#pragma unroll
    for (int mask = 1; mask <= 8; mask <<= 1)
#pragma unroll
      for (int r = 0; r < 8; ++r) rs[r] += __shfl_xor(rs[r], mask, 32);
#pragma unroll
    for (int r = 0; r < 8; ++r) l[r] = l[r] * corr[r] + rs[r];

    // P: C-layout -> A-layout via per-wave LDS bounce (same-wave LDS is
    // in-order; s_wait_dscnt 0 makes stored data visible to all lanes).
#pragma unroll
    for (int t = 0; t < 2; ++t)
#pragma unroll
      for (int r = 0; r < 8; ++r)
        pl[(r + hi * 8) * 36 + t * 16 + l16] = (_Float16)s[t][r];
    asm volatile("s_wait_dscnt 0" ::: "memory");
    HFrag pf;
    {
      const uint32_t* p = (const uint32_t*)(pl + l16 * 36);
#pragma unroll
      for (int v = 0; v < 8; ++v) pf.u[v] = p[a_kd(v, hi) >> 1];
    }

    // Rescale accumulators, then O += P * V (V^T layout -> contiguous B frags).
#pragma unroll
    for (int t = 0; t < 4; ++t)
#pragma unroll
      for (int r = 0; r < 8; ++r) acc[t][r] *= corr[r];
#pragma unroll
    for (int t = 0; t < 4; ++t) {
      HFrag vf;
      const int d = t * 16 + l16;
      const uint32_t* p = (const uint32_t*)(vp + (size_t)d * 1024 + j);
#pragma unroll
      for (int v = 0; v < 8; ++v) vf.u[v] = p[v + hi * 8];
      acc[t] = WMMA_F16(pf.v, vf.v, acc[t]);
    }
  }

  // Finalize: divide by softmax denominator, write f16 [B,N,C].
#pragma unroll
  for (int r = 0; r < 8; ++r) l[r] = 1.0f / l[r];
#pragma unroll
  for (int t = 0; t < 4; ++t) {
    const int d = t * 16 + l16;
#pragma unroll
    for (int r = 0; r < 8; ++r) {
      const int row = qbase + r + hi * 8;
      aout[((size_t)batch * 1024 + row) * 768 + head * 64 + d] =
          (_Float16)(acc[t][r] * l[r]);
    }
  }
}

// ---------------------------------------------------------------------------
// Kernel 3: output projection.  aout[4096,768] (f16) @ proj_w[768,768] + b -> f32.
// A-tile staged with CDNA5 async copy (global_load_async_to_lds_b128).
// ---------------------------------------------------------------------------
__global__ __launch_bounds__(256) void proj_kernel(
    const _Float16* __restrict__ a, const float* __restrict__ w,
    const float* __restrict__ bias, float* __restrict__ out) {
  __shared__ _Float16 As[128 * 40];   // stride 40 halfs = 80B -> 16B-aligned rows
  __shared__ _Float16 Bs[128 * 36];
  const int tid = threadIdx.x;
  const int lane = tid & 31, wave = tid >> 5;
  const int wm = wave >> 2, wn = wave & 3;
  const int l16 = lane & 15, hi = lane >> 4;
  const int m0 = blockIdx.y * 128, n0 = blockIdx.x * 128;

  v8f acc[4][2];
#pragma unroll
  for (int mi = 0; mi < 4; ++mi)
#pragma unroll
    for (int ni = 0; ni < 2; ++ni)
#pragma unroll
      for (int r = 0; r < 8; ++r) acc[mi][ni][r] = 0.0f;

  for (int k0 = 0; k0 < 768; k0 += 32) {
    __syncthreads();
    // A tile: 128x32 f16, async DMA straight into LDS (no VGPR bounce).
#pragma unroll
    for (int it = 0; it < 2; ++it) {
      const int idx = tid + it * 256;          // 0..511 chunks of 8 halfs (16B)
      const int row = idx >> 2, kc = (idx & 3) << 3;
      const _Float16* g = a + (size_t)(m0 + row) * 768 + k0 + kc;
      const uint32_t ldsoff = (uint32_t)(uintptr_t)(As + row * 40 + kc);
      asm volatile("global_load_async_to_lds_b128 %0, %1, off"
                   :: "v"(ldsoff), "v"(g) : "memory");
    }
    // B tile: 32x128 f32 -> f16, stored transposed [n][k]
#pragma unroll
    for (int it = 0; it < 4; ++it) {
      const int idx = tid + it * 256;
      const int kk = idx >> 5, nc = (idx & 31) << 2;
      const float4 f = *(const float4*)(w + (size_t)(k0 + kk) * 768 + n0 + nc);
      Bs[(nc + 0) * 36 + kk] = (_Float16)f.x;
      Bs[(nc + 1) * 36 + kk] = (_Float16)f.y;
      Bs[(nc + 2) * 36 + kk] = (_Float16)f.z;
      Bs[(nc + 3) * 36 + kk] = (_Float16)f.w;
    }
    asm volatile("s_wait_asynccnt 0" ::: "memory");
    __syncthreads();

    HFrag af[4], bf[2];
#pragma unroll
    for (int mi = 0; mi < 4; ++mi) {
      const uint32_t* p = (const uint32_t*)(As + (wm * 64 + mi * 16 + l16) * 40);
#pragma unroll
      for (int v = 0; v < 8; ++v) af[mi].u[v] = p[a_kd(v, hi) >> 1];
    }
#pragma unroll
    for (int ni = 0; ni < 2; ++ni) {
      const uint32_t* p = (const uint32_t*)(Bs + (wn * 32 + ni * 16 + l16) * 36);
#pragma unroll
      for (int v = 0; v < 8; ++v) bf[ni].u[v] = p[v + hi * 8];
    }
#pragma unroll
    for (int mi = 0; mi < 4; ++mi)
#pragma unroll
      for (int ni = 0; ni < 2; ++ni)
        acc[mi][ni] = WMMA_F16(af[mi].v, bf[ni].v, acc[mi][ni]);
  }

#pragma unroll
  for (int mi = 0; mi < 4; ++mi) {
#pragma unroll
    for (int ni = 0; ni < 2; ++ni) {
      const int col = n0 + wn * 32 + ni * 16 + l16;
      const float bv = bias[col];
#pragma unroll
      for (int r = 0; r < 8; ++r) {
        const int row = m0 + wm * 64 + mi * 16 + r + hi * 8;
        out[(size_t)row * 768 + col] = acc[mi][ni][r] + bv;
      }
    }
  }
}

// ---------------------------------------------------------------------------
extern "C" void kernel_launch(void* const* d_in, const int* in_sizes, int n_in,
                              void* d_out, int out_size, void* d_ws, size_t ws_size,
                              hipStream_t stream) {
  (void)in_sizes; (void)n_in; (void)out_size; (void)ws_size;
  const float* x      = (const float*)d_in[0];
  const float* abias  = (const float*)d_in[1];
  const float* qkv_w  = (const float*)d_in[2];
  const float* qkv_b  = (const float*)d_in[3];
  const float* proj_w = (const float*)d_in[4];
  const float* proj_b = (const float*)d_in[5];
  float* out = (float*)d_out;

  const size_t QKV = (size_t)4 * 12 * 1024 * 64;  // 3,145,728 elems per buffer
  _Float16* qb   = (_Float16*)d_ws;
  _Float16* kbuf = qb + QKV;
  _Float16* vtb  = kbuf + QKV;
  _Float16* aout = vtb + QKV;   // total ~25.2 MB of workspace

  qkv_kernel <<<dim3(18, 32), 256, 0, stream>>>(x, qkv_w, qkv_b, qb, kbuf, vtb);
  attn_kernel<<<dim3(16, 48), 128, 0, stream>>>(qb, kbuf, vtb, abias, aout);
  proj_kernel<<<dim3(6, 32),  256, 0, stream>>>(aout, proj_w, proj_b, out);
}
